// TFLOPLoss_64854006170237
// MI455X (gfx1250) — compile-verified, matrix operand order checked by reading, not run on hardware
//
#include <hip/hip_runtime.h>
#include <hip/hip_bf16.h>
#include <math.h>

// Problem constants (match reference): B=8, L=1024, D=512, V=64, K=8
#define B_ 8
#define L_ 1024
#define D_ 512
#define V_ 64
#define K_ 8
#define INV_TEMP 10.0f
#define EPS_ 1e-10f

typedef __attribute__((ext_vector_type(16))) __bf16 v16bf;
typedef __attribute__((ext_vector_type(8)))  float  v8f;

// ---------------- helpers ----------------

// f32 -> bf16 with round-to-nearest-even (bit trick; avoids relying on __bf16 arithmetic)
__device__ __forceinline__ unsigned short f2bf(float f) {
  unsigned int u = __float_as_uint(f);
  unsigned int r = u + 0x7FFFu + ((u >> 16) & 1u);
  return (unsigned short)(r >> 16);
}
__device__ __forceinline__ unsigned int pack2bf(float a, float b) {
  return (unsigned int)f2bf(a) | ((unsigned int)f2bf(b) << 16);
}

// Load one 16x16x32 bf16 WMMA fragment (16 bf16 = 8 VGPRs) from an LDS row.
// Per ISA 7.12.2 (16-bit A 16x32): VGPR0-3 hold K=base..base+7, VGPR4-7 hold
// K=base+16..base+23 (with base = kc*32 + (lane>=16)*8). Two ds_load_b128.
__device__ __forceinline__ v16bf load_frag(const unsigned short* rowp, int koff) {
  union { uint4 q[2]; v16bf v; } x;
  x.q[0] = *(const uint4*)(rowp + koff);
  x.q[1] = *(const uint4*)(rowp + koff + 16);
  return x.v;
}

__device__ __forceinline__ float wave_red_add16(float v) {   // butterfly over 16-lane half
  v += __shfl_xor(v, 1); v += __shfl_xor(v, 2);
  v += __shfl_xor(v, 4); v += __shfl_xor(v, 8);
  return v;
}
__device__ __forceinline__ float wave_red_add32(float v) {   // full wave32 butterfly
  v += __shfl_xor(v, 1); v += __shfl_xor(v, 2); v += __shfl_xor(v, 4);
  v += __shfl_xor(v, 8); v += __shfl_xor(v, 16);
  return v;
}
__device__ __forceinline__ float wave_red_max32(float v) {
  v = fmaxf(v, __shfl_xor(v, 1));  v = fmaxf(v, __shfl_xor(v, 2));
  v = fmaxf(v, __shfl_xor(v, 4));  v = fmaxf(v, __shfl_xor(v, 8));
  v = fmaxf(v, __shfl_xor(v, 16));
  return v;
}

// ---------------- 1) classification loss ----------------
// One wave per (b,l) row of V=64 logits. Accumulate sum(per_tok*valid), sum(valid).
__global__ __launch_bounds__(256) void cls_kernel(const float* __restrict__ logits,
                                                  const int* __restrict__ tok,
                                                  float* __restrict__ acc) {
  const int lane = threadIdx.x & 31;
  const int row  = blockIdx.x * 8 + (threadIdx.x >> 5);   // row in [0, B*L)
  const float* x = logits + (size_t)row * V_;
  float x0 = x[lane], x1 = x[lane + 32];
  float m  = wave_red_max32(fmaxf(x0, x1));
  float e  = __expf(x0 - m) + __expf(x1 - m);
  float lse = __logf(wave_red_add32(e)) + m;
  float sx  = wave_red_add32(x0 + x1);
  int t = tok[row];                                       // uniform across wave
  float xt = __shfl((t < 32) ? x0 : x1, t & 31);
  float nll    = lse - xt;
  float smooth = lse - sx * (1.0f / V_);
  float per    = 0.9f * nll + 0.1f * smooth;
  bool valid   = !(t == 0 || t == 1 || t == 2 || t == 3); // SPECIAL_IDS
  if (lane == 0) {
    atomicAdd(&acc[0], valid ? per : 0.0f);
    atomicAdd(&acc[1], valid ? 1.0f : 0.0f);
  }
}

// ---------------- 2) fused bf16-WMMA GEMM + masked-exp row reduce -> den ----------------
// MT=4 row-tiles (64 l-rows) per block: cuts tag_proj L2 traffic 4x (1GB -> 256MB)
// and amortizes each staged B fragment over 4 WMMAs.
#define GW  4      // waves per block
#define MT  4      // 16-row l-tiles per block
#define LDP 520    // padded LDS row (520 shorts = 1040B; row stride rotates banks by 4 dwords)

__global__ __launch_bounds__(32 * GW) void den_gemm_kernel(
    const float* __restrict__ box, const float* __restrict__ tag,
    const unsigned char* __restrict__ dtm, float* __restrict__ den) {
  __shared__ unsigned short sA[MT * 16][LDP];   // box rows (bf16), shared by block (~65KB)
  __shared__ unsigned short sB[GW][16][LDP];    // per-wave tag tile (bf16)      (~65KB)
  __shared__ float denAcc[MT * 16];

  const int b    = blockIdx.x >> 4;             // L_/(16*MT) == 16 l-blocks per batch
  const int l0   = (blockIdx.x & 15) << 6;      // 64 rows per block
  const int tid  = threadIdx.x;
  const int wave = tid >> 5, lane = tid & 31;
  const int l15  = lane & 15, hi = lane >> 4;

  for (int i = tid; i < MT * 16; i += 32 * GW) denAcc[i] = 0.0f;

  // Stage A rows: 64 x 512 f32 -> bf16 in LDS (coalesced float4 global loads)
  const float4* Ab = (const float4*)(box + ((size_t)b * L_ + l0) * D_);
  for (int i = tid; i < MT * 16 * (D_ / 4); i += 32 * GW) {
    float4 v = Ab[i];
    int r = i >> 7, c = (i & 127) << 2;         // 128 float4 per row
    *(uint2*)&sA[r][c] = make_uint2(pack2bf(v.x, v.y), pack2bf(v.z, v.w));
  }
  __syncthreads();

  const unsigned char* dtmb = dtm + (size_t)b * (2 * L_) + L_;  // data_tag_mask[:, L:]

  for (int tt = wave; tt < L_ / 16; tt += GW) {          // uniform (64 % GW == 0)
    const int t0 = tt << 4;
    // Stage this wave's tag tile into its private LDS region
    const float4* Bb = (const float4*)(tag + ((size_t)b * L_ + t0) * D_);
    for (int i = lane; i < 16 * (D_ / 4); i += 32) {
      float4 v = Bb[i];
      int r = i >> 7, c = (i & 127) << 2;
      *(uint2*)&sB[wave][r][c] = make_uint2(pack2bf(v.x, v.y), pack2bf(v.z, v.w));
    }
    // Intra-wave producer->consumer: LDS stores complete before cross-lane reads.
    asm volatile("s_wait_dscnt 0" ::: "memory");

    v8f acc[MT] = {};
    #pragma unroll
    for (int kc = 0; kc < D_ / 32; ++kc) {               // K = 512 -> 16 iterations
      const int koff = kc * 32 + hi * 8;
      v16bf bfrag = load_frag(&sB[wave][l15][0], koff);  // B: N = lane&15
      #pragma unroll
      for (int mt = 0; mt < MT; ++mt) {                  // 4 WMMAs share one B fragment
        v16bf afrag = load_frag(&sA[mt * 16 + l15][0], koff);
        acc[mt] = __builtin_amdgcn_wmma_f32_16x16x32_bf16(
            /*neg_a=*/false, afrag, /*neg_b=*/false, bfrag,
            /*c_mod=*/(short)0, acc[mt], /*reuse_a=*/false, /*reuse_b=*/false);
      }
    }
    // C layout: VGPR r, lane -> M = r + 8*hi, N = lane&15 (t index within tile)
    const float mN = dtmb[t0 + l15] ? 1.0f : 0.0f;
    #pragma unroll
    for (int mt = 0; mt < MT; ++mt) {
      #pragma unroll
      for (int r = 0; r < 8; ++r) {
        float e = mN * __expf(acc[mt][r] * INV_TEMP);    // raw exp, matches reference
        float s = wave_red_add16(e);                     // sum over N (16 lanes per half)
        if (l15 == 0) atomicAdd(&denAcc[mt * 16 + r + 8 * hi], s);  // ds_add_f32
      }
    }
  }
  __syncthreads();
  for (int i = tid; i < MT * 16; i += 32 * GW)
    den[(size_t)b * L_ + l0 + i] = denAcc[i];
}

// ---------------- 3) pointer loss (recompute K=8 target dots; avoids 32MB logits) ----------------
__global__ __launch_bounds__(256) void ptr_kernel(
    const int* __restrict__ boxidx, const unsigned char* __restrict__ dtm,
    const float* __restrict__ box, const float* __restrict__ tag,
    const float* __restrict__ den, float* __restrict__ acc) {
  const int lane = threadIdx.x & 31;
  const int row  = blockIdx.x * 8 + (threadIdx.x >> 5);  // row = b*L + l
  const int b    = row >> 10;
  const int* bi  = boxidx + (size_t)row * K_;
  const unsigned char* dtmb = dtm + (size_t)b * (2 * L_) + L_;
  const float* a = box + (size_t)row * D_;
  const float dl = __logf(den[row] + EPS_);
  float sum = 0.0f, cnt = 0.0f;
  bool alive = true;
  for (int k = 0; k < K_; ++k) {
    int idx = bi[k];                                     // uniform scalar per wave
    alive = alive && (idx != -1);                        // cumprod validity
    int rc = min(max(idx - L_, 0), L_ - 1);
    bool mm = alive && (dtmb[rc] != 0);                  // wave-uniform branch
    if (mm) {
      const float* c = tag + ((size_t)b * L_ + rc) * D_;
      float d = 0.0f;
      for (int dd = lane; dd < D_; dd += 32) d += a[dd] * c[dd];
      d = wave_red_add32(d);
      sum += dl - d * INV_TEMP;
      cnt += 1.0f;
    }
  }
  if (lane == 0) { atomicAdd(&acc[2], sum); atomicAdd(&acc[3], cnt); }
}

// ---------------- 4) empty-pointer BCE loss ----------------
__global__ __launch_bounds__(256) void empty_kernel(
    const float* __restrict__ eproj, const float* __restrict__ tag,
    const unsigned char* __restrict__ att, const unsigned char* __restrict__ dtm,
    float* __restrict__ acc) {
  __shared__ float sE[D_];
  __shared__ float sSum, sCnt;
  const int b = blockIdx.x, tid = threadIdx.x, lane = tid & 31, wave = tid >> 5;
  if (tid == 0) { sSum = 0.0f; sCnt = 0.0f; }
  for (int i = tid; i < D_; i += 256) sE[i] = eproj[(size_t)b * D_ + i];
  __syncthreads();
  const unsigned char* attb = att + (size_t)b * (2 * L_) + L_;
  const unsigned char* dtmb = dtm + (size_t)b * (2 * L_) + L_;
  float ls = 0.0f, lc = 0.0f;
  for (int t = wave; t < L_; t += 8) {
    const float* c = tag + ((size_t)b * L_ + t) * D_;
    float d = 0.0f;
    for (int dd = lane; dd < D_; dd += 32) d += sE[dd] * c[dd];
    d = wave_red_add32(d);
    bool av   = attb[t] != 0;
    float tgt = (av && !dtmb[t]) ? 1.0f : 0.0f;
    float sp  = fmaxf(d, 0.0f) + log1pf(__expf(-fabsf(d)));   // stable softplus
    ls += av ? (sp - d * tgt) : 0.0f;
    lc += av ? 1.0f : 0.0f;
  }
  if (lane == 0) { atomicAdd(&sSum, ls); atomicAdd(&sCnt, lc); }
  __syncthreads();
  if (tid == 0) atomicAdd(&acc[4], (sSum / fmaxf(sCnt, 1.0f)) * (1.0f / B_));
}

// ---------------- 5) span-contrastive (shared by row & col) ----------------
// One block per (b,j) row; 4 columns/thread kept in registers -> each 32MB array read once.
__global__ __launch_bounds__(256) void span_kernel(
    const float* __restrict__ sim, const float* __restrict__ coef,
    float* __restrict__ tot, float* __restrict__ cnt) {
  __shared__ float red[256];
  const int row = blockIdx.x;                 // b*L + j
  const int jd  = row & (L_ - 1);             // diagonal column
  const int tid = threadIdx.x;
  const float* sp = sim  + (size_t)row * L_;
  const float* cp = coef + (size_t)row * L_;
  float s[4], w[4];
  float lmax = -3.4e38f;
  #pragma unroll
  for (int q = 0; q < 4; ++q) {
    int c = tid + q * 256;
    s[q] = sp[c] * INV_TEMP;
    float cf = cp[c];
    w[q] = cf > 0.0f ? cf : 0.0f;
    lmax = fmaxf(lmax, s[q]);
  }
  red[tid] = lmax; __syncthreads();
  for (int st = 128; st > 0; st >>= 1) { if (tid < st) red[tid] = fmaxf(red[tid], red[tid + st]); __syncthreads(); }
  float rmax = red[0]; __syncthreads();

  float esum = 0.0f;
  #pragma unroll
  for (int q = 0; q < 4; ++q) {
    int c = tid + q * 256;
    s[q] -= rmax;
    float e = __expf(s[q]);
    if (c != jd) esum += e;                   // exclude diagonal from denominator
  }
  red[tid] = esum; __syncthreads();
  for (int st = 128; st > 0; st >>= 1) { if (tid < st) red[tid] += red[tid + st]; __syncthreads(); }
  float logden = __logf(red[0] + EPS_); __syncthreads();

  float lnum = 0.0f, lw = 0.0f;
  #pragma unroll
  for (int q = 0; q < 4; ++q) { lnum += w[q] * (logden - s[q]); lw += w[q]; }
  red[tid] = lw; __syncthreads();
  for (int st = 128; st > 0; st >>= 1) { if (tid < st) red[tid] += red[tid + st]; __syncthreads(); }
  float wsum = red[0]; __syncthreads();
  red[tid] = lnum; __syncthreads();
  for (int st = 128; st > 0; st >>= 1) { if (tid < st) red[tid] += red[tid + st]; __syncthreads(); }
  if (tid == 0 && wsum > 0.0f) {              // wsum>0 <=> any(coef>0)
    atomicAdd(tot, red[0] / (wsum + EPS_));
    atomicAdd(cnt, 1.0f);
  }
}

// ---------------- 6) finalize ----------------
__global__ void finalize_kernel(const float* __restrict__ acc, float* __restrict__ out) {
  float cls = acc[0] / fmaxf(acc[1], 1.0f);
  float ptr = acc[3] > 0.0f ? acc[2] / fmaxf(acc[3], 1.0f) : 0.0f;
  float emp = acc[4];
  float row = acc[6] > 0.0f ? acc[5] / acc[6] : 0.0f;
  float col = acc[8] > 0.0f ? acc[7] / acc[8] : 0.0f;
  out[0] = cls + ptr + emp + 0.5f * row + 0.5f * col;
  out[1] = cls; out[2] = ptr; out[3] = emp; out[4] = row; out[5] = col;
}

// ---------------- launcher ----------------
extern "C" void kernel_launch(void* const* d_in, const int* in_sizes, int n_in,
                              void* d_out, int out_size, void* d_ws, size_t ws_size,
                              hipStream_t stream) {
  const int*   token_ids   = (const int*)d_in[0];
  const int*   box_indices = (const int*)d_in[1];
  const unsigned char* data_tag_mask  = (const unsigned char*)d_in[2];  // bool (1B)
  const unsigned char* attention_mask = (const unsigned char*)d_in[3];  // bool (1B)
  const float* tag_logits = (const float*)d_in[4];
  const float* box_proj   = (const float*)d_in[5];
  const float* tag_proj   = (const float*)d_in[6];
  const float* empty_proj = (const float*)d_in[7];
  const float* row_sim    = (const float*)d_in[8];
  const float* col_sim    = (const float*)d_in[9];
  const float* row_coef   = (const float*)d_in[10];
  const float* col_coef   = (const float*)d_in[11];

  float* acc = (float*)d_ws;        // [0..8] scalar accumulators
  float* den = acc + 16;            // B*L floats (fully written by den_gemm_kernel)
  hipMemsetAsync(d_ws, 0, 16 * sizeof(float), stream);   // capturable async memset

  cls_kernel<<<B_ * L_ / 8, 256, 0, stream>>>(tag_logits, token_ids, acc);
  den_gemm_kernel<<<B_ * (L_ / (16 * MT)), 32 * GW, 0, stream>>>(box_proj, tag_proj, data_tag_mask, den);
  ptr_kernel<<<B_ * L_ / 8, 256, 0, stream>>>(box_indices, data_tag_mask, box_proj, tag_proj, den, acc);
  empty_kernel<<<B_, 256, 0, stream>>>(empty_proj, tag_proj, attention_mask, data_tag_mask, acc);
  span_kernel<<<B_ * L_, 256, 0, stream>>>(row_sim, row_coef, &acc[5], &acc[6]);
  span_kernel<<<B_ * L_, 256, 0, stream>>>(col_sim, col_coef, &acc[7], &acc[8]);
  finalize_kernel<<<1, 1, 0, stream>>>(acc, (float*)d_out);
}
